// GraphAttentionLayer_2138893714143
// MI455X (gfx1250) — matrix operand order repaired and checked by reading, not compiled, not run on hardware
//
#include <hip/hip_runtime.h>
#include <stdint.h>

#define ALPHA_F 0.2f
#define MASK_FILL_F (-9.0e15f)

typedef __attribute__((ext_vector_type(16))) __bf16 v16bf;
typedef __attribute__((ext_vector_type(8)))  float  v8f;
typedef __attribute__((ext_vector_type(4)))  unsigned int v4u;
typedef __attribute__((ext_vector_type(8)))  int    v8i;
typedef __attribute__((ext_vector_type(4)))  int    v4i;

union BFrag { unsigned short s[16]; uint4 q[2]; v16bf v; };

// ---------------------------------------------------------------------------
// TDM: issue a 2D tile load (global -> LDS). Descriptor per CDNA5 ISA §8.
// ---------------------------------------------------------------------------
static __device__ __forceinline__ void tdm_load_2d(
        const void* gaddr, uint32_t lds_byte_addr,
        uint32_t dsz_code,                 // 0=1B 1=2B 2=4B
        uint32_t tensor_d0, uint32_t tensor_d1,
        uint32_t tile_d0, uint32_t tile_d1,
        uint64_t stride0_elems) {
    uint64_t ga = (uint64_t)(uintptr_t)gaddr;
    v4u g0;
    g0.x = 1u;                                   // count=1, user descriptor
    g0.y = lds_byte_addr;                        // LDS byte offset
    g0.z = (uint32_t)ga;                         // global_addr[31:0]
    g0.w = (uint32_t)((ga >> 32) & 0x01FFFFFFull) | (2u << 30); // ga[56:32] | type=2
    v8i g1;
    g1.s0 = (int)(dsz_code << 16);               // wg_mask=0 | data_size
    g1.s1 = (int)((tensor_d0 & 0xFFFFu) << 16);  // tensor_dim0[15:0]
    g1.s2 = (int)((tensor_d0 >> 16) | ((tensor_d1 & 0xFFFFu) << 16));
    g1.s3 = (int)((tensor_d1 >> 16) | (tile_d0 << 16));
    g1.s4 = (int)(tile_d1 & 0xFFFFu);            // tile_dim1 | tile_dim2=0
    g1.s5 = (int)(uint32_t)stride0_elems;        // dim0_stride[31:0]
    g1.s6 = (int)(uint32_t)(stride0_elems >> 32);// dim0_stride[47:32] | dim1_stride lo=0
    g1.s7 = 0;
    v4i z4 = {};
#if defined(__clang_major__) && (__clang_major__ >= 23)
    v8i z8 = {};
    __builtin_amdgcn_tensor_load_to_lds(g0, g1, z4, z4, z8, 0);
#else
    __builtin_amdgcn_tensor_load_to_lds(g0, g1, z4, z4, 0);
#endif
}

// ---------------------------------------------------------------------------
// Kernel 0: W (512x256 f32, row-major) -> Wt bf16 (256x512), K-contiguous rows
// ---------------------------------------------------------------------------
__global__ void __launch_bounds__(256) k_transposeW(
        const float* __restrict__ W, __bf16* __restrict__ WtBf) {
    int idx = blockIdx.x * 256 + threadIdx.x;    // 0 .. 131071
    int n = idx >> 9;                            // feature 0..255
    int k = idx & 511;                           // K 0..511
    WtBf[idx] = (__bf16)W[k * 256 + n];          // hw v_cvt bf16
}

// ---------------------------------------------------------------------------
// Kernel 1: Wh = h @ W via v_wmma_f32_16x16x32_bf16.
// One 16x16 tile per wave; K=512 in 16 steps.
// Writes Wh (f32 row-major) and WhT (bf16 256x8192, K-contiguous for attn B).
// ---------------------------------------------------------------------------
__global__ void __launch_bounds__(256) k_gemm_wh(
        const float* __restrict__ h, const __bf16* __restrict__ WtBf,
        float* __restrict__ Wh, __bf16* __restrict__ WhT) {
    int w    = threadIdx.x >> 5;
    int lane = threadIdx.x & 31;
    int t    = blockIdx.x * 8 + w;               // 8192 tiles: 512 x 16
    int m0   = (t >> 4) << 4;
    int n0   = (t & 15) << 4;
    int lanen  = lane & 15;
    int hi     = lane >> 4;
    int kgroup = hi << 3;                        // A: 0 or 8
    int khalf  = hi << 4;                        // unused: 0 or 16

    const float*  hrow = h    + (size_t)(m0 + lanen) * 512;
    const __bf16* brow = WtBf + (size_t)(n0 + lanen) * 512;

    v8f acc = {};
    for (int kk = 0; kk < 512; kk += 32) {
        // A fragment: elems 0..7 -> K=kk+kgroup+0..7 ; 8..15 -> K=kk+kgroup+16..23
        float4 f0 = *(const float4*)(hrow + kk + kgroup);
        float4 f1 = *(const float4*)(hrow + kk + kgroup + 4);
        float4 f2 = *(const float4*)(hrow + kk + kgroup + 16);
        float4 f3 = *(const float4*)(hrow + kk + kgroup + 20);
        float fa[16] = { f0.x,f0.y,f0.z,f0.w, f1.x,f1.y,f1.z,f1.w,
                         f2.x,f2.y,f2.z,f2.w, f3.x,f3.y,f3.z,f3.w };
        v16bf av;
        #pragma unroll
        for (int e = 0; e < 16; ++e) av[e] = (__bf16)fa[e];   // hw cvt (packable)
        // B fragment: lane col = n0+lanen, elems 0..7 -> K=kk+kgroup+0..7 ; 8..15 -> K=kk+kgroup+16..23
        BFrag b;
        b.q[0] = *(const uint4*)(brow + kk + kgroup);
        b.q[1] = *(const uint4*)(brow + kk + kgroup + 16);
        acc = __builtin_amdgcn_wmma_f32_16x16x32_bf16(
                  false, av, false, b.v, (short)0, acc, false, false);
    }
    int col = n0 + lanen;
    #pragma unroll
    for (int r = 0; r < 8; ++r) {
        int m = m0 + r + 8 * hi;
        float vv = acc[r];
        Wh[(size_t)m * 256 + col]   = vv;
        WhT[(size_t)col * 8192 + m] = (__bf16)vv;
    }
}

// ---------------------------------------------------------------------------
// Kernel 2: src = Wh@a1 , dst = Wh@a2  (one row per thread)
// ---------------------------------------------------------------------------
__global__ void __launch_bounds__(256) k_srcdst(
        const float* __restrict__ Wh, const float* __restrict__ a,
        float* __restrict__ srcv, float* __restrict__ dstv) {
    int r = blockIdx.x * 256 + threadIdx.x;      // 0..8191
    const float4* wr = (const float4*)(Wh + (size_t)r * 256);
    const float4* a1 = (const float4*)(a);
    const float4* a2 = (const float4*)(a + 256);
    float s1 = 0.f, s2 = 0.f;
    #pragma unroll 8
    for (int i = 0; i < 64; ++i) {
        float4 wv = wr[i], x = a1[i], y = a2[i];
        s1 += wv.x*x.x + wv.y*x.y + wv.z*x.z + wv.w*x.w;
        s2 += wv.x*y.x + wv.y*y.y + wv.z*y.z + wv.w*y.w;
    }
    srcv[r] = s1;
    dstv[r] = s2;
}

// ---------------------------------------------------------------------------
// Kernel 3: fused masked-attention (flash-softmax) + P@Wh + ELU.
// 128 blocks x 256 thr; block = 64 rows; keys stream in 32-wide chunks
// staged to LDS by the Tensor Data Mover, double buffered.
// wave w: row tile rt=w>>1 (16 rows), feature half (w&1)*128 (8x 16-col tiles)
// ---------------------------------------------------------------------------
__global__ void __launch_bounds__(256) k_attn(
        const int* __restrict__ adj, const __bf16* __restrict__ WhT,
        const float* __restrict__ srcv, const float* __restrict__ dstv,
        float* __restrict__ out) {
    __shared__ int            adjS[2][64 * 32];   // 2 x 8 KB
    __shared__ unsigned short whS [2][256 * 32];  // 2 x 16 KB (bf16 payload)
    __shared__ float          dstS[2][32];        // 2 x 128 B

    const int tid   = threadIdx.x;
    const int w     = tid >> 5;
    const int lane  = tid & 31;
    const int rt    = w >> 1;
    const int fbase = (w & 1) * 128;
    const int lanen = lane & 15;
    const int hi    = lane >> 4;
    const int kgroup = hi << 3;                   // 0 / 8
    const int khalf  = hi << 4;                   // 0 / 16
    const int rowl  = rt * 16 + lanen;
    const int rowg  = blockIdx.x * 64 + rowl;

    const float my_src = srcv[rowg];
    float m_run = -3.0e38f, l_run = 0.0f;
    v8f acc[8];
    #pragma unroll
    for (int f = 0; f < 8; ++f) acc[f] = (v8f){};

    const int NCH = 8192 / 32;
    const int* adj_blk = adj + (uint64_t)blockIdx.x * 524288ull; // 64*8192

    if (w == 0) {
        tdm_load_2d(adj_blk,  (uint32_t)(uintptr_t)&adjS[0][0], 2, 8192, 8192, 32,  64, 8192);
        tdm_load_2d(WhT,      (uint32_t)(uintptr_t)&whS [0][0], 1, 8192,  256, 32, 256, 8192);
        tdm_load_2d(dstv,     (uint32_t)(uintptr_t)&dstS[0][0], 2, 8192,    1, 32,   0, 8192);
    }

    for (int ci = 0; ci < NCH; ++ci) {
        const int cur = ci & 1;
        if (w == 0) __builtin_amdgcn_s_wait_tensorcnt(0);
        __syncthreads();                                   // buffers[cur] ready
        if (w == 0 && (ci + 1) < NCH) {
            const int c2 = (ci + 1) * 32;
            tdm_load_2d(adj_blk + c2, (uint32_t)(uintptr_t)&adjS[cur ^ 1][0], 2, 8192, 8192, 32,  64, 8192);
            tdm_load_2d(WhT + c2,     (uint32_t)(uintptr_t)&whS [cur ^ 1][0], 1, 8192,  256, 32, 256, 8192);
            tdm_load_2d(dstv + c2,    (uint32_t)(uintptr_t)&dstS[cur ^ 1][0], 2, 8192,    1, 32,   0, 8192);
        }

        // ---- scores for this lane's 16 (row, key) elements (A-frag order) --
        const int* arow = &adjS[cur][rowl * 32];
        int4   a0 = *(const int4*)(arow + kgroup);
        int4   a1 = *(const int4*)(arow + kgroup + 4);
        int4   a2 = *(const int4*)(arow + kgroup + 16);
        int4   a3 = *(const int4*)(arow + kgroup + 20);
        float4 d0 = *(const float4*)(&dstS[cur][kgroup]);
        float4 d1 = *(const float4*)(&dstS[cur][kgroup + 4]);
        float4 d2 = *(const float4*)(&dstS[cur][kgroup + 16]);
        float4 d3 = *(const float4*)(&dstS[cur][kgroup + 20]);
        int   am[16] = { a0.x,a0.y,a0.z,a0.w, a1.x,a1.y,a1.z,a1.w,
                         a2.x,a2.y,a2.z,a2.w, a3.x,a3.y,a3.z,a3.w };
        float dd[16] = { d0.x,d0.y,d0.z,d0.w, d1.x,d1.y,d1.z,d1.w,
                         d2.x,d2.y,d2.z,d2.w, d3.x,d3.y,d3.z,d3.w };
        float s[16];
        #pragma unroll
        for (int e = 0; e < 16; ++e) {
            float t = my_src + dd[e];
            t = fmaxf(t, ALPHA_F * t);                     // leaky relu (alpha<1)
            s[e] = (am[e] > 0) ? t : MASK_FILL_F;          // adjacency mask
        }

        // ---- online softmax: row split across lanes l and l^16 ------------
        float mc = s[0];
        #pragma unroll
        for (int e = 1; e < 16; ++e) mc = fmaxf(mc, s[e]);
        mc = fmaxf(mc, __shfl_xor(mc, 16));
        const float m_new = fmaxf(m_run, mc);
        const float scale = __expf(m_run - m_new);

        v16bf pv; float psum = 0.0f;
        #pragma unroll
        for (int e = 0; e < 16; ++e) {
            float p = __expf(s[e] - m_new);
            psum += p;
            pv[e] = (__bf16)p;                             // hw cvt (packable)
        }
        psum += __shfl_xor(psum, 16);
        l_run = l_run * scale + psum;
        m_run = m_new;

        // ---- rescale accumulators (per C-fragment row) --------------------
        v8f sc;
        #pragma unroll
        for (int r = 0; r < 8; ++r) sc[r] = __shfl(scale, r + 8 * hi);
        #pragma unroll
        for (int f = 0; f < 8; ++f) acc[f] *= sc;

        // ---- P (16x32) @ Wh-chunk (32x16 per feature tile) ----------------
        #pragma unroll
        for (int f = 0; f < 8; ++f) {
            const unsigned short* bptr =
                &whS[cur][(fbase + f * 16 + lanen) * 32 + kgroup];
            BFrag b;
            b.q[0] = *(const uint4*)(bptr);
            b.q[1] = *(const uint4*)(bptr + 16);
            acc[f] = __builtin_amdgcn_wmma_f32_16x16x32_bf16(
                         false, pv, false, b.v, (short)0, acc[f], false, false);
        }
        __syncthreads();                                   // done with buffers[cur]
    }

    // ---- normalize, ELU, store --------------------------------------------
    float linv[8];
    #pragma unroll
    for (int r = 0; r < 8; ++r) linv[r] = 1.0f / __shfl(l_run, r + 8 * hi);
    #pragma unroll
    for (int f = 0; f < 8; ++f) {
        #pragma unroll
        for (int r = 0; r < 8; ++r) {
            float x = acc[f][r] * linv[r];
            x = (x > 0.0f) ? x : (__expf(x) - 1.0f);       // ELU
            size_t row = (size_t)blockIdx.x * 64 + rt * 16 + r + 8 * hi;
            out[row * 256 + fbase + f * 16 + lanen] = x;
        }
    }
}

// ---------------------------------------------------------------------------
extern "C" void kernel_launch(void* const* d_in, const int* in_sizes, int n_in,
                              void* d_out, int out_size, void* d_ws, size_t ws_size,
                              hipStream_t stream) {
    (void)in_sizes; (void)n_in; (void)out_size; (void)ws_size;
    const float* h   = (const float*)d_in[0];
    const int*   adj = (const int*)  d_in[1];
    const float* W   = (const float*)d_in[2];
    const float* a   = (const float*)d_in[3];
    float* out = (float*)d_out;

    char* ws = (char*)d_ws;
    float*  Wh    = (float*)(ws);                              // 8 MB
    __bf16* WhT   = (__bf16*)(ws + 8388608);                   // 4 MB
    __bf16* WtBf  = (__bf16*)(ws + 8388608 + 4194304);         // 256 KB
    float*  srcv  = (float*)(ws + 8388608 + 4194304 + 262144); // 32 KB
    float*  dstv  = srcv + 8192;                               // 32 KB

    k_transposeW<<<512, 256, 0, stream>>>(W, WtBf);
    k_gemm_wh  <<<1024, 256, 0, stream>>>(h, WtBf, Wh, WhT);
    k_srcdst   <<<32,   256, 0, stream>>>(Wh, a, srcv, dstv);
    k_attn     <<<128,  256, 0, stream>>>(adj, WhT, srcv, dstv, out);
}
